// MoELayer_71038759076630
// MI455X (gfx1250) — matrix operand order repaired and checked by reading, not compiled, not run on hardware
//
#include <hip/hip_runtime.h>
#include <stdint.h>

#define NTOK 4096      // B*S
#define DIMK 2048      // model dim
#define NEXP 8
#define HDIM 2730      // expert hidden
#define HPAD 2752      // padded to multiple of 64
#define LDA  40        // LDS row stride in bf16 elements (80B, bank-conflict free)

typedef __attribute__((ext_vector_type(16))) __bf16 v16bf;
typedef __attribute__((ext_vector_type(8)))  float  v8f;
typedef __attribute__((ext_vector_type(4)))  int    v4i;

union FragBF { uint4 q[2]; v16bf v; };
static_assert(sizeof(v16bf) == 32, "v16bf must be 32B");

#define AS1 __attribute__((address_space(1)))
#define AS3 __attribute__((address_space(3)))

#if __has_builtin(__builtin_amdgcn_global_load_async_to_lds_b128) && \
    __has_builtin(__builtin_amdgcn_s_wait_asynccnt)
#define USE_ASYNC 1
#else
#define USE_ASYNC 0
#endif

#if USE_ASYNC
__device__ __forceinline__ void async_b128(const void* g, void* l) {
    __builtin_amdgcn_global_load_async_to_lds_b128(
        (AS1 v4i*)g, (AS3 v4i*)l, 0, 0);
}
#endif

__device__ __forceinline__ uint16_t f2bf(float f) {
    union { float f; uint32_t u; } c; c.f = f;
    uint32_t u = c.u;
    u += 0x7fffu + ((u >> 16) & 1u);   // round-to-nearest-even
    return (uint16_t)(u >> 16);
}

// ---------------- fp32 -> bf16 cast (x) ----------------
__global__ void k_cast_bf16(const float* __restrict__ src,
                            uint16_t* __restrict__ dst, int n) {
    int i = blockIdx.x * blockDim.x + threadIdx.x;
    int stride = gridDim.x * blockDim.x;
    for (; i < n; i += stride) dst[i] = f2bf(src[i]);
}

// ---------------- fp32 -> bf16 with zero padding (weights) ----------------
__global__ void k_cast_pad(const float* __restrict__ src, uint16_t* __restrict__ dst,
                           int srows, int scols, int drows, int dcols) {
    int i = blockIdx.x * blockDim.x + threadIdx.x;
    int stride = gridDim.x * blockDim.x;
    int total = drows * dcols;
    for (; i < total; i += stride) {
        int r = i / dcols, c = i - r * dcols;
        dst[i] = (r < srows && c < scols) ? f2bf(src[(size_t)r * scols + c])
                                          : (uint16_t)0;
    }
}

// ---------------- router: softmax -> top-2 -> renormalized gates ----------------
__global__ void k_router(const float* __restrict__ x, const float* __restrict__ wrw,
                         float* __restrict__ gates) {
    int gid  = blockIdx.x * blockDim.x + threadIdx.x;
    int tok  = gid >> 5;
    int lane = gid & 31;
    if (tok >= NTOK) return;
    const float* xr = x + (size_t)tok * DIMK;
    float lg[NEXP];
    #pragma unroll
    for (int e = 0; e < NEXP; ++e) {
        const float* w = wrw + (size_t)e * DIMK;
        float s = 0.f;
        for (int i = lane; i < DIMK; i += 32) s += xr[i] * w[i];
        #pragma unroll
        for (int off = 16; off; off >>= 1) s += __shfl_xor(s, off, 32);
        lg[e] = s;
    }
    float m = lg[0];
    #pragma unroll
    for (int e = 1; e < NEXP; ++e) m = fmaxf(m, lg[e]);
    float p[NEXP], sum = 0.f;
    #pragma unroll
    for (int e = 0; e < NEXP; ++e) { p[e] = __expf(lg[e] - m); sum += p[e]; }
    float inv = 1.f / sum;
    #pragma unroll
    for (int e = 0; e < NEXP; ++e) p[e] *= inv;
    int i1 = 0;
    #pragma unroll
    for (int e = 1; e < NEXP; ++e) if (p[e] > p[i1]) i1 = e;
    int i2 = (i1 == 0) ? 1 : 0;
    #pragma unroll
    for (int e = 0; e < NEXP; ++e) if (e != i1 && p[e] > p[i2]) i2 = e;
    float denom = p[i1] + p[i2];
    if (lane < NEXP) {
        float g = (lane == i1) ? (p[i1] / denom)
                : (lane == i2) ? (p[i2] / denom) : 0.f;
        gates[(size_t)tok * NEXP + lane] = g;
    }
}

// ---------------- up-projection: Hbuf = silu(X W1^T) * (X W3^T) ----------------
// block tile: 128 tokens x 64 hidden cols, BK=32, 8 waves (4 in M, 2 in N),
// each wave: 2x2 WMMA 16x16 tiles for BOTH U and V (A-fragments reused).
__global__ void __launch_bounds__(256)
k_swiglu_up(const uint16_t* __restrict__ X, const uint16_t* __restrict__ W1,
            const uint16_t* __restrict__ W3, uint16_t* __restrict__ Hbuf) {
    __shared__ uint16_t Xs [2][128 * LDA];
    __shared__ uint16_t W1s[2][ 64 * LDA];
    __shared__ uint16_t W3s[2][ 64 * LDA];

    const int t    = threadIdx.x;
    const int lane = t & 31, wave = t >> 5;
    const int wm   = wave & 3;        // wave row tile (32 tokens)
    const int wn   = wave >> 2;       // wave col tile (32 hidden)
    const int hBase   = blockIdx.x * 64;
    const int tokBase = blockIdx.y * 128;

    // global staging assignments
    const int xr = t >> 1, xh = t & 1;     // X: 128 rows x two 32B chunks
    const int wr2 = t >> 2, wc = t & 3;    // W: 64 rows x four 16B chunks
    const uint16_t* Xg  = X  + (size_t)(tokBase + xr ) * DIMK + xh * 16;
    const uint16_t* W1g = W1 + (size_t)(hBase   + wr2) * DIMK + wc * 8;
    const uint16_t* W3g = W3 + (size_t)(hBase   + wr2) * DIMK + wc * 8;

    v8f accU[2][2] = {};
    v8f accV[2][2] = {};

    // prologue: stage k=0 tile
#if USE_ASYNC
    async_b128(Xg,      &Xs [0][xr  * LDA + xh * 16    ]);
    async_b128(Xg + 8,  &Xs [0][xr  * LDA + xh * 16 + 8]);
    async_b128(W1g,     &W1s[0][wr2 * LDA + wc * 8     ]);
    async_b128(W3g,     &W3s[0][wr2 * LDA + wc * 8     ]);
    __builtin_amdgcn_s_wait_asynccnt(0);
#else
    {
        uint4 x0 = ((const uint4*)Xg)[0], x1 = ((const uint4*)Xg)[1];
        uint4 a1 = *(const uint4*)W1g,    a3 = *(const uint4*)W3g;
        *(uint4*)&Xs [0][xr  * LDA + xh * 16    ] = x0;
        *(uint4*)&Xs [0][xr  * LDA + xh * 16 + 8] = x1;
        *(uint4*)&W1s[0][wr2 * LDA + wc * 8     ] = a1;
        *(uint4*)&W3s[0][wr2 * LDA + wc * 8     ] = a3;
    }
#endif
    __syncthreads();

    const int rsel = lane & 15, hsel = lane >> 4;
    const int ksteps = DIMK / 32;
    int buf = 0;
    for (int kk = 0; kk < ksteps; ++kk) {
#if USE_ASYNC
        if (kk + 1 < ksteps) {                     // async-stage next tile
            const int nb = buf ^ 1;
            const uint16_t* xp = Xg + (kk + 1) * 32;
            async_b128(xp,     &Xs [nb][xr  * LDA + xh * 16    ]);
            async_b128(xp + 8, &Xs [nb][xr  * LDA + xh * 16 + 8]);
            async_b128(W1g + (kk + 1) * 32, &W1s[nb][wr2 * LDA + wc * 8]);
            async_b128(W3g + (kk + 1) * 32, &W3s[nb][wr2 * LDA + wc * 8]);
        }
#else
        uint4 nx0, nx1, nw1, nw3;
        if (kk + 1 < ksteps) {                     // issue next-tile loads early
            const uint16_t* xp = Xg + (kk + 1) * 32;
            nx0 = ((const uint4*)xp)[0];
            nx1 = ((const uint4*)xp)[1];
            nw1 = *(const uint4*)(W1g + (kk + 1) * 32);
            nw3 = *(const uint4*)(W3g + (kk + 1) * 32);
        }
        if (kk + 2 < ksteps) {                     // L2 prefetch two tiles ahead
            __builtin_prefetch(W1g + (kk + 2) * 32, 0, 0);
            __builtin_prefetch(W3g + (kk + 2) * 32, 0, 0);
        }
#endif

        FragBF a[2], b1[2], b3[2];
        #pragma unroll
        for (int mt = 0; mt < 2; ++mt) {           // A: 16x32 bf16 fragment
            const uint16_t* p = &Xs[buf][(wm * 32 + mt * 16 + rsel) * LDA];
            a[mt].q[0] = *(const uint4*)(p + hsel * 16);
            a[mt].q[1] = *(const uint4*)(p + hsel * 16 + 8);
        }
        #pragma unroll
        for (int nt = 0; nt < 2; ++nt) {           // B: 32x16 bf16 fragments
            const uint16_t* p1 = &W1s[buf][(wn * 32 + nt * 16 + rsel) * LDA + hsel * 16];
            b1[nt].q[0] = ((const uint4*)p1)[0];
            b1[nt].q[1] = ((const uint4*)p1)[1];
            const uint16_t* p3 = &W3s[buf][(wn * 32 + nt * 16 + rsel) * LDA + hsel * 16];
            b3[nt].q[0] = ((const uint4*)p3)[0];
            b3[nt].q[1] = ((const uint4*)p3)[1];
        }
        #pragma unroll
        for (int mt = 0; mt < 2; ++mt)
            #pragma unroll
            for (int nt = 0; nt < 2; ++nt) {
                accU[mt][nt] = __builtin_amdgcn_wmma_f32_16x16x32_bf16(
                    false, a[mt].v, false, b1[nt].v, (short)0, accU[mt][nt], false, false);
                accV[mt][nt] = __builtin_amdgcn_wmma_f32_16x16x32_bf16(
                    false, a[mt].v, false, b3[nt].v, (short)0, accV[mt][nt], false, false);
            }

#if USE_ASYNC
        __builtin_amdgcn_s_wait_asynccnt(0);       // this wave's tile is in LDS
#else
        if (kk + 1 < ksteps) {
            const int nb = buf ^ 1;
            *(uint4*)&Xs [nb][xr  * LDA + xh * 16    ] = nx0;
            *(uint4*)&Xs [nb][xr  * LDA + xh * 16 + 8] = nx1;
            *(uint4*)&W1s[nb][wr2 * LDA + wc * 8     ] = nw1;
            *(uint4*)&W3s[nb][wr2 * LDA + wc * 8     ] = nw3;
        }
#endif
        __syncthreads();
        buf ^= 1;
    }

    // epilogue: h = silu(u) * v, store bf16
    #pragma unroll
    for (int mt = 0; mt < 2; ++mt)
        #pragma unroll
        for (int nt = 0; nt < 2; ++nt)
            #pragma unroll
            for (int g = 0; g < 8; ++g) {
                float u = accU[mt][nt][g];
                float h = (u / (1.f + __expf(-u))) * accV[mt][nt][g];
                int tok = tokBase + wm * 32 + mt * 16 + g + 8 * hsel;
                int hc  = hBase   + wn * 32 + nt * 16 + rsel;
                Hbuf[(size_t)tok * HPAD + hc] = f2bf(h);
            }
}

// ---------------- down-projection: out (+)= gate * (Hbuf W2^T) ----------------
// block tile: 128 tokens x 128 dim cols, BK=32, 8 waves (4 in M, 2 in N),
// each wave: 2x4 WMMA tiles (32x64) -> 8 WMMAs per 12 ds_load_b128.
__global__ void __launch_bounds__(256)
k_down_proj(const uint16_t* __restrict__ Hb, const uint16_t* __restrict__ W2,
            const float* __restrict__ gates, int eidx, int accumulate,
            float* __restrict__ out) {
    __shared__ uint16_t As[2][128 * LDA];
    __shared__ uint16_t Bs[2][128 * LDA];

    const int t    = threadIdx.x;
    const int lane = t & 31, wave = t >> 5;
    const int wm   = wave & 3;     // 32-token row tile
    const int wn   = wave >> 2;    // 64-col tile
    const int dBase   = blockIdx.x * 128;
    const int tokBase = blockIdx.y * 128;

    const int sr = t >> 1, sh = t & 1;   // staging: 128 rows x two 32B chunks
    const uint16_t* Ag = Hb + (size_t)(tokBase + sr) * HPAD + sh * 16;
    const uint16_t* Bg = W2 + (size_t)(dBase   + sr) * HPAD + sh * 16;

    v8f acc[2][4] = {};

#if USE_ASYNC
    async_b128(Ag,     &As[0][sr * LDA + sh * 16    ]);
    async_b128(Ag + 8, &As[0][sr * LDA + sh * 16 + 8]);
    async_b128(Bg,     &Bs[0][sr * LDA + sh * 16    ]);
    async_b128(Bg + 8, &Bs[0][sr * LDA + sh * 16 + 8]);
    __builtin_amdgcn_s_wait_asynccnt(0);
#else
    {
        uint4 a0 = ((const uint4*)Ag)[0], a1 = ((const uint4*)Ag)[1];
        uint4 b0 = ((const uint4*)Bg)[0], b1 = ((const uint4*)Bg)[1];
        *(uint4*)&As[0][sr * LDA + sh * 16    ] = a0;
        *(uint4*)&As[0][sr * LDA + sh * 16 + 8] = a1;
        *(uint4*)&Bs[0][sr * LDA + sh * 16    ] = b0;
        *(uint4*)&Bs[0][sr * LDA + sh * 16 + 8] = b1;
    }
#endif
    __syncthreads();

    const int rsel = lane & 15, hsel = lane >> 4;
    const int ksteps = HPAD / 32;   // 86
    int buf = 0;
    for (int kk = 0; kk < ksteps; ++kk) {
#if USE_ASYNC
        if (kk + 1 < ksteps) {
            const int nb = buf ^ 1;
            const uint16_t* ap = Ag + (kk + 1) * 32;
            const uint16_t* bp = Bg + (kk + 1) * 32;
            async_b128(ap,     &As[nb][sr * LDA + sh * 16    ]);
            async_b128(ap + 8, &As[nb][sr * LDA + sh * 16 + 8]);
            async_b128(bp,     &Bs[nb][sr * LDA + sh * 16    ]);
            async_b128(bp + 8, &Bs[nb][sr * LDA + sh * 16 + 8]);
        }
#else
        uint4 na0, na1, nb0, nb1;
        if (kk + 1 < ksteps) {
            const uint16_t* ap = Ag + (kk + 1) * 32;
            const uint16_t* bp = Bg + (kk + 1) * 32;
            na0 = ((const uint4*)ap)[0];
            na1 = ((const uint4*)ap)[1];
            nb0 = ((const uint4*)bp)[0];
            nb1 = ((const uint4*)bp)[1];
        }
        if (kk + 2 < ksteps) __builtin_prefetch(Bg + (kk + 2) * 32, 0, 0);
#endif

        FragBF a[2], b[4];
        #pragma unroll
        for (int mt = 0; mt < 2; ++mt) {
            const uint16_t* p = &As[buf][(wm * 32 + mt * 16 + rsel) * LDA];
            a[mt].q[0] = *(const uint4*)(p + hsel * 16);
            a[mt].q[1] = *(const uint4*)(p + hsel * 16 + 8);
        }
        #pragma unroll
        for (int nt = 0; nt < 4; ++nt) {
            const uint16_t* p = &Bs[buf][(wn * 64 + nt * 16 + rsel) * LDA + hsel * 16];
            b[nt].q[0] = ((const uint4*)p)[0];
            b[nt].q[1] = ((const uint4*)p)[1];
        }
        #pragma unroll
        for (int mt = 0; mt < 2; ++mt)
            #pragma unroll
            for (int nt = 0; nt < 4; ++nt)
                acc[mt][nt] = __builtin_amdgcn_wmma_f32_16x16x32_bf16(
                    false, a[mt].v, false, b[nt].v, (short)0, acc[mt][nt], false, false);

#if USE_ASYNC
        __builtin_amdgcn_s_wait_asynccnt(0);
#else
        if (kk + 1 < ksteps) {
            const int nb = buf ^ 1;
            *(uint4*)&As[nb][sr * LDA + sh * 16    ] = na0;
            *(uint4*)&As[nb][sr * LDA + sh * 16 + 8] = na1;
            *(uint4*)&Bs[nb][sr * LDA + sh * 16    ] = nb0;
            *(uint4*)&Bs[nb][sr * LDA + sh * 16 + 8] = nb1;
        }
#endif
        __syncthreads();
        buf ^= 1;
    }

    #pragma unroll
    for (int mt = 0; mt < 2; ++mt)
        #pragma unroll
        for (int nt = 0; nt < 4; ++nt)
            #pragma unroll
            for (int g = 0; g < 8; ++g) {
                int tok = tokBase + wm * 32 + mt * 16 + g + 8 * hsel;
                int d   = dBase   + wn * 64 + nt * 16 + rsel;
                float gate = (eidx < 0) ? 1.f : gates[(size_t)tok * NEXP + eidx];
                float val  = gate * acc[mt][nt][g];
                size_t o = (size_t)tok * DIMK + d;
                if (accumulate) out[o] += val;
                else            out[o]  = val;
            }
}

// ---------------- host launcher ----------------
extern "C" void kernel_launch(void* const* d_in, const int* in_sizes, int n_in,
                              void* d_out, int out_size, void* d_ws, size_t ws_size,
                              hipStream_t stream) {
    const float* x   = (const float*)d_in[0];
    const float* sw1 = (const float*)d_in[1];
    const float* sw2 = (const float*)d_in[2];
    const float* sw3 = (const float*)d_in[3];
    const float* ew1 = (const float*)d_in[4];
    const float* ew2 = (const float*)d_in[5];
    const float* ew3 = (const float*)d_in[6];
    const float* wrw = (const float*)d_in[7];
    float* out = (float*)d_out;

    char* ws = (char*)d_ws;
    size_t off = 0;
    auto alloc = [&](size_t bytes) -> void* {
        void* p = ws + off;
        off = (off + bytes + 255) & ~((size_t)255);
        return p;
    };
    uint16_t* xbf   = (uint16_t*)alloc((size_t)NTOK * DIMK * 2);
    float*    gates = (float*)   alloc((size_t)NTOK * NEXP * 4);
    uint16_t* Hbuf  = (uint16_t*)alloc((size_t)NTOK * HPAD * 2);
    uint16_t* w1bf  = (uint16_t*)alloc((size_t)HPAD * DIMK * 2);
    uint16_t* w3bf  = (uint16_t*)alloc((size_t)HPAD * DIMK * 2);
    uint16_t* w2bf  = (uint16_t*)alloc((size_t)DIMK * HPAD * 2);
    (void)in_sizes; (void)n_in; (void)out_size; (void)ws_size;

    k_cast_bf16<<<1024, 256, 0, stream>>>(x, xbf, NTOK * DIMK);
    k_router<<<(NTOK * 32) / 256, 256, 0, stream>>>(x, wrw, gates);

    dim3 g1(HPAD / 64,  NTOK / 128);   // 43 x 32
    dim3 g2(DIMK / 128, NTOK / 128);   // 16 x 32
    for (int e = 0; e < 9; ++e) {
        const float *pw1, *pw2, *pw3;
        if (e == 0) { pw1 = sw1; pw2 = sw2; pw3 = sw3; }
        else {
            pw1 = ew1 + (size_t)(e - 1) * HDIM * DIMK;
            pw2 = ew2 + (size_t)(e - 1) * DIMK * HDIM;
            pw3 = ew3 + (size_t)(e - 1) * HDIM * DIMK;
        }
        k_cast_pad<<<2048, 256, 0, stream>>>(pw1, w1bf, HDIM, DIMK, HPAD, DIMK);
        k_cast_pad<<<2048, 256, 0, stream>>>(pw3, w3bf, HDIM, DIMK, HPAD, DIMK);
        k_cast_pad<<<2048, 256, 0, stream>>>(pw2, w2bf, DIMK, HDIM, DIMK, HPAD);
        k_swiglu_up<<<g1, 256, 0, stream>>>(xbf, w1bf, w3bf, Hbuf);
        k_down_proj<<<g2, 256, 0, stream>>>(Hbuf, w2bf, gates, e - 1, e > 0 ? 1 : 0, out);
    }
}
